// GraphSAGEEdgeClassifier_77025943486768
// MI455X (gfx1250) — matrix operand order repaired and checked by reading, not compiled, not run on hardware
//
#include <hip/hip_runtime.h>
#include <hip/hip_bf16.h>

typedef __attribute__((ext_vector_type(16))) __bf16 v16bf;
typedef __attribute__((ext_vector_type(8)))  float  v8f;

#define HID 128
// Padded LDS row strides (in bf16 elements) to avoid 16-way bank conflicts on
// fragment reads (unpadded 128*2B = 256B = exactly 64 banks -> all rows same bank).
#define LDA128 136   // 68 dwords per row: consecutive rows shift 4 banks
#define LDA256 264   // 132 dwords per row

// ---------------------------------------------------------------------------
// Helpers
// ---------------------------------------------------------------------------

__device__ __forceinline__ unsigned pack_bf16(float x, float y) {
  // lowers to v_cvt_pk_bf16_f32
  __bf16 lo = (__bf16)x, hi = (__bf16)y;
  unsigned short ulo = __builtin_bit_cast(unsigned short, lo);
  unsigned short uhi = __builtin_bit_cast(unsigned short, hi);
  return (unsigned)ulo | ((unsigned)uhi << 16);
}

// A-matrix 16x32 bf16 fragment from a bf16 row (LDS), rowptr = row base + 32*kb.
// Layout (ISA 7.12.2): lane&15 = M; VGPR i<4 -> K = 2i       + 8*(lane>=16)
//                                   VGPR i>=4 -> K = 16+2(i-4) + 8*(lane>=16)
__device__ __forceinline__ v16bf load_a_frag_bf16(const __bf16* rowptr, int kgroup) {
  v16bf a;
#pragma unroll
  for (int i = 0; i < 8; ++i) {
    int k0 = ((i < 4) ? 2 * i : 16 + 2 * (i - 4)) + 8 * kgroup;
    a[2 * i]     = rowptr[k0];
    a[2 * i + 1] = rowptr[k0 + 1];   // pairs contiguous -> 4B ds reads
  }
  return a;
}

// B-matrix 32x16 bf16 fragment from a column-major weight (colptr = W^T column
// base, length K). lane&15 = N; VGPR j<4 -> K = 2j + 8*(lane>=16), j>=4 -> K = 16+2(j-4) + 8*(lane>=16).
// 16 bf16 per lane in two contiguous runs of 8 -> wide global loads, L0-resident weights.
__device__ __forceinline__ v16bf load_b_frag(const __bf16* __restrict__ colptr,
                                             int kb, int kgroup) {
  const __bf16* p = colptr + 32 * kb + 8 * kgroup;
  v16bf b;
#pragma unroll
  for (int j = 0; j < 8; ++j) {
    b[2 * j]     = p[2 * j + ((j < 4) ? 0 : 8)];
    b[2 * j + 1] = p[2 * j + 1 + ((j < 4) ? 0 : 8)];
  }
  return b;
}

// ---------------------------------------------------------------------------
// Sparse side: degree histogram + scatter-add aggregation (L2-resident atomics)
// ---------------------------------------------------------------------------

__global__ void deg_kernel(const int* __restrict__ dst, float* __restrict__ deg, int E) {
  int t = blockIdx.x * blockDim.x + threadIdx.x;
  if (t < E) atomicAdd(deg + dst[t], 1.0f);
}

// one wave32 per edge; each lane does 4 feature atomics (coalesced 512B rows)
__global__ void scatter_kernel(const float* __restrict__ x,
                               const int* __restrict__ src,
                               const int* __restrict__ dst,
                               float* __restrict__ agg, int E) {
  long long t = (long long)blockIdx.x * blockDim.x + threadIdx.x;
  int e = (int)(t >> 5);
  if (e >= E) return;
  int h = (int)(t & 31);
  const float* xr = x + (size_t)src[e] * HID;
  float* ar = agg + (size_t)dst[e] * HID;
#pragma unroll
  for (int i = 0; i < 4; ++i)
    atomicAdd(ar + h + 32 * i, xr[h + 32 * i]);
}

// ---------------------------------------------------------------------------
// Weight transpose + f32 -> bf16 convert:  Wt[j*K + k] = W[k*Nout + j] (pad j)
// ---------------------------------------------------------------------------

__global__ void wconv_kernel(const float* __restrict__ W, __bf16* __restrict__ Wt,
                             int K, int Nout, int NoutPad) {
  int t = blockIdx.x * blockDim.x + threadIdx.x;
  if (t >= K * NoutPad) return;
  int j = t / K, k = t - j * K;
  float v = (j < Nout) ? W[(size_t)k * Nout + j] : 0.0f;
  Wt[(size_t)j * K + k] = (__bf16)v;
}

// ---------------------------------------------------------------------------
// Fused SAGE layer: out = relu(mean @ Wl + x @ Wr + b).
// Stage 1: 256 threads cooperatively build the block's two bf16 A-tiles
//          (mean = agg * 1/deg formed inline) in LDS -- each element converted
//          ONCE instead of once per wave.
// Stage 2: 8 waves, wave w computes cols [16w,16w+16) via 8 WMMAs from LDS.
// ---------------------------------------------------------------------------

__global__ __launch_bounds__(256)
void sage_gemm_kernel(const float* __restrict__ agg, const float* __restrict__ deg,
                      const float* __restrict__ x,
                      const __bf16* __restrict__ Wlt, const __bf16* __restrict__ Wrt,
                      const float* __restrict__ bias, float* __restrict__ out, int M) {
  __shared__ unsigned ldsAu[2][16 * (LDA128 / 2)];   // mean tile, x tile (bf16)

  const int lane   = threadIdx.x & 31;
  const int wave   = threadIdx.x >> 5;   // column tile 0..7
  const int kgroup = lane >> 4;
  const int hrow   = lane & 15;
  const int r0     = blockIdx.x * 16;

  // ---- stage 1: f32 -> bf16 tile staging (float4 granularity) ----
#pragma unroll
  for (int it = 0; it < 2; ++it) {
    int idx  = threadIdx.x + it * 256;   // 512 float4 slots = 16 rows x 32
    int row  = idx >> 5;
    int k4   = (idx & 31) * 4;
    int grow = r0 + row;
    if (grow >= M) grow = M - 1;
    float ri = 1.0f / fmaxf(deg[grow], 1.0f);
    const float4 va = *(const float4*)(agg + (size_t)grow * HID + k4);
    const float4 vx = *(const float4*)(x   + (size_t)grow * HID + k4);
    int ui = row * (LDA128 / 2) + (k4 >> 1);
    uint2 pa, px;
    pa.x = pack_bf16(va.x * ri, va.y * ri);
    pa.y = pack_bf16(va.z * ri, va.w * ri);
    px.x = pack_bf16(vx.x, vx.y);
    px.y = pack_bf16(vx.z, vx.w);
    *(uint2*)(&ldsAu[0][ui]) = pa;
    *(uint2*)(&ldsAu[1][ui]) = px;
  }
  __syncthreads();

  // ---- stage 2: WMMA from LDS tiles ----
  const int col = wave * 16 + hrow;
  const __bf16* a1row = (const __bf16*)ldsAu[0] + hrow * LDA128;
  const __bf16* a2row = (const __bf16*)ldsAu[1] + hrow * LDA128;
  const __bf16* wl    = Wlt + (size_t)col * HID;
  const __bf16* wr    = Wrt + (size_t)col * HID;

  v8f c;
  const float bv = bias[col];
#pragma unroll
  for (int i = 0; i < 8; ++i) c[i] = bv;

#pragma unroll
  for (int kb = 0; kb < 4; ++kb) {       // K = 128 = 4 x 32
    v16bf a1 = load_a_frag_bf16(a1row + 32 * kb, kgroup);
    v16bf b1 = load_b_frag(wl, kb, kgroup);
    c = __builtin_amdgcn_wmma_f32_16x16x32_bf16(false, a1, false, b1, (short)0, c, false, false);
    v16bf a2 = load_a_frag_bf16(a2row + 32 * kb, kgroup);
    v16bf b2 = load_b_frag(wr, kb, kgroup);
    c = __builtin_amdgcn_wmma_f32_16x16x32_bf16(false, a2, false, b2, (short)0, c, false, false);
  }

#pragma unroll
  for (int i = 0; i < 8; ++i) {          // c[i] -> row (i + 8*kgroup), col
    int rg = r0 + i + 8 * kgroup;
    if (rg < M) out[(size_t)rg * HID + col] = fmaxf(c[i], 0.0f);
  }
}

// ---------------------------------------------------------------------------
// Fused edge classifier.
// Stage 1: cooperative gather-concat [x[qs] | x[qd]] -> 16x256 bf16 LDS tile
//          (each gathered element converted once per block).
// Stage 2: MLP1 (K=256) via WMMA, relu -> 16x128 bf16 LDS tile.
// Stage 3: wave 0 does MLP2 (K=128, Nout padded 8->16) straight to d_out.
// ---------------------------------------------------------------------------

__global__ __launch_bounds__(256)
void edge_mlp_kernel(const float* __restrict__ h,
                     const int* __restrict__ qs, const int* __restrict__ qd,
                     const __bf16* __restrict__ W1t, const float* __restrict__ b1,
                     const __bf16* __restrict__ W2t, const float* __restrict__ b2,
                     float* __restrict__ out, int Eq) {
  __shared__ unsigned ldsGu[16 * (LDA256 / 2)];      // gathered edge features
  __shared__ unsigned ldsHu[16 * (LDA128 / 2)];      // hidden activations

  const int lane   = threadIdx.x & 31;
  const int wave   = threadIdx.x >> 5;
  const int kgroup = lane >> 4;
  const int hrow   = lane & 15;
  const int r0     = blockIdx.x * 16;

  // ---- stage 1: gather + f32 -> bf16 concat tile ----
#pragma unroll
  for (int it = 0; it < 4; ++it) {
    int idx  = threadIdx.x + it * 256;   // 1024 float4 slots = 16 rows x 64
    int row  = idx >> 6;
    int k4   = (idx & 63) * 4;
    int grow = r0 + row;
    if (grow >= Eq) grow = Eq - 1;
    int node = (k4 < HID) ? qs[grow] : qd[grow];
    int koff = (k4 < HID) ? k4 : (k4 - HID);
    const float4 v = *(const float4*)(h + (size_t)node * HID + koff);
    int ui = row * (LDA256 / 2) + (k4 >> 1);
    uint2 p;
    p.x = pack_bf16(v.x, v.y);
    p.y = pack_bf16(v.z, v.w);
    *(uint2*)(&ldsGu[ui]) = p;
  }
  __syncthreads();

  // ---- stage 2: MLP1, wave w -> cols [16w,16w+16) of 128 ----
  const int col = wave * 16 + hrow;
  const __bf16* arow = (const __bf16*)ldsGu + hrow * LDA256;
  const __bf16* w1   = W1t + (size_t)col * 256;

  v8f c;
  const float bv = b1[col];
#pragma unroll
  for (int i = 0; i < 8; ++i) c[i] = bv;

#pragma unroll
  for (int kb = 0; kb < 8; ++kb) {       // K = 256
    v16bf a = load_a_frag_bf16(arow + 32 * kb, kgroup);
    v16bf b = load_b_frag(w1, kb, kgroup);
    c = __builtin_amdgcn_wmma_f32_16x16x32_bf16(false, a, false, b, (short)0, c, false, false);
  }

  __bf16* ldsH = (__bf16*)ldsHu;
#pragma unroll
  for (int i = 0; i < 8; ++i)            // relu -> bf16 staging tile
    ldsH[(i + 8 * kgroup) * LDA128 + col] = (__bf16)fmaxf(c[i], 0.0f);

  __syncthreads();

  // ---- stage 3: MLP2 on wave 0 (wave-uniform branch: EXEC stays full) ----
  if (wave == 0) {
    const __bf16* w2 = W2t + (size_t)hrow * HID;   // col = hrow (16 padded cols)
    v8f c2;
    const float b2v = (hrow < 8) ? b2[hrow] : 0.0f;
#pragma unroll
    for (int i = 0; i < 8; ++i) c2[i] = b2v;

#pragma unroll
    for (int kb = 0; kb < 4; ++kb) {     // K = 128 from LDS
      v16bf a = load_a_frag_bf16(ldsH + hrow * LDA128 + 32 * kb, kgroup);
      v16bf b = load_b_frag(w2, kb, kgroup);
      c2 = __builtin_amdgcn_wmma_f32_16x16x32_bf16(false, a, false, b, (short)0, c2, false, false);
    }

#pragma unroll
    for (int i = 0; i < 8; ++i) {
      int rg = r0 + i + 8 * kgroup;
      if (rg < Eq && hrow < 8) out[(size_t)rg * 8 + hrow] = c2[i];
    }
  }
}

// ---------------------------------------------------------------------------
// Host launcher
// ---------------------------------------------------------------------------

extern "C" void kernel_launch(void* const* d_in, const int* in_sizes, int n_in,
                              void* d_out, int out_size, void* d_ws, size_t ws_size,
                              hipStream_t stream) {
  const float* node_emb = (const float*)d_in[0];
  const float* Wl0 = (const float*)d_in[1];
  const float* bl0 = (const float*)d_in[2];
  const float* Wr0 = (const float*)d_in[3];
  const float* Wl1 = (const float*)d_in[4];
  const float* bl1 = (const float*)d_in[5];
  const float* Wr1 = (const float*)d_in[6];
  const float* W1  = (const float*)d_in[7];
  const float* b1  = (const float*)d_in[8];
  const float* W2  = (const float*)d_in[9];
  const float* b2  = (const float*)d_in[10];
  const int* edge_index = (const int*)d_in[11];
  const int* edge_q     = (const int*)d_in[12];

  const int N  = in_sizes[0] / HID;
  const int E  = in_sizes[11] / 2;
  const int Eq = in_sizes[12] / 2;

  // ---- carve scratch (~155 MB working set; resident in the 192 MB L2) ----
  char* ws = (char*)d_ws;
  size_t off = 0;
  auto carve = [&](size_t bytes) -> void* {
    void* p = ws + off;
    off = (off + bytes + 255) & ~(size_t)255;
    return p;
  };
  float*  agg  = (float*)carve((size_t)N * HID * sizeof(float));
  float*  h0   = (float*)carve((size_t)N * HID * sizeof(float));
  float*  h1   = (float*)carve((size_t)N * HID * sizeof(float));
  float*  deg  = (float*)carve((size_t)N * sizeof(float));
  __bf16* Wl0t = (__bf16*)carve(HID * HID * sizeof(__bf16));
  __bf16* Wr0t = (__bf16*)carve(HID * HID * sizeof(__bf16));
  __bf16* Wl1t = (__bf16*)carve(HID * HID * sizeof(__bf16));
  __bf16* Wr1t = (__bf16*)carve(HID * HID * sizeof(__bf16));
  __bf16* W1t  = (__bf16*)carve(256 * HID * sizeof(__bf16));
  __bf16* W2t  = (__bf16*)carve(16 * HID * sizeof(__bf16));
  (void)ws_size; (void)n_in; (void)out_size;

  // ---- weight transpose/convert (tiny, one-time per call) ----
  {
    int n = HID * HID, g = (n + 255) / 256;
    wconv_kernel<<<g, 256, 0, stream>>>(Wl0, Wl0t, HID, HID, HID);
    wconv_kernel<<<g, 256, 0, stream>>>(Wr0, Wr0t, HID, HID, HID);
    wconv_kernel<<<g, 256, 0, stream>>>(Wl1, Wl1t, HID, HID, HID);
    wconv_kernel<<<g, 256, 0, stream>>>(Wr1, Wr1t, HID, HID, HID);
    int n1 = 256 * HID;
    wconv_kernel<<<(n1 + 255) / 256, 256, 0, stream>>>(W1, W1t, 256, HID, HID);
    int n2 = HID * 16;
    wconv_kernel<<<(n2 + 255) / 256, 256, 0, stream>>>(W2, W2t, HID, 8, 16);
  }

  const int* esrc = edge_index;
  const int* edst = edge_index + E;

  // ---- degree (depends only on dst: compute once) ----
  hipMemsetAsync(deg, 0, (size_t)N * sizeof(float), stream);
  deg_kernel<<<(E + 255) / 256, 256, 0, stream>>>(edst, deg, E);

  const unsigned scat_blocks = (unsigned)(((long long)E * 32 + 255) / 256);
  const unsigned gemm_blocks = (unsigned)((N + 15) / 16);

  // ---- layer 0 ----
  hipMemsetAsync(agg, 0, (size_t)N * HID * sizeof(float), stream);
  scatter_kernel<<<scat_blocks, 256, 0, stream>>>(node_emb, esrc, edst, agg, E);
  sage_gemm_kernel<<<gemm_blocks, 256, 0, stream>>>(agg, deg, node_emb, Wl0t, Wr0t, bl0, h0, N);

  // ---- layer 1 ----
  hipMemsetAsync(agg, 0, (size_t)N * HID * sizeof(float), stream);
  scatter_kernel<<<scat_blocks, 256, 0, stream>>>(h0, esrc, edst, agg, E);
  sage_gemm_kernel<<<gemm_blocks, 256, 0, stream>>>(agg, deg, h0, Wl1t, Wr1t, bl1, h1, N);

  // ---- fused edge classifier ----
  edge_mlp_kernel<<<(unsigned)((Eq + 15) / 16), 256, 0, stream>>>(
      h1, edge_q, edge_q + Eq, W1t, b1, W2t, b2, (float*)d_out, Eq);
}
